// TransformerEncoder_86457691669082
// MI455X (gfx1250) — matrix operand order repaired and checked by reading, not compiled, not run on hardware
//
#include <hip/hip_runtime.h>
#include <hip/hip_bf16.h>

typedef __attribute__((ext_vector_type(16))) _Float16 v16h;
typedef __attribute__((ext_vector_type(8)))  float    v8f;

#define WMMA16(a, b, c) \
    __builtin_amdgcn_wmma_f32_16x16x32_f16(false, (a), false, (b), (short)0, (c), false, false)

// ---------------- problem constants ----------------
constexpr int Bb  = 2;
constexpr int Ss  = 2048;
constexpr int DM  = 1024;
constexpr int HIDc = 256;
constexpr int DFF = 512;
constexpr int Hh  = 4;
constexpr int DHh = 64;
constexpr int Mrows = Bb * Ss;      // 4096

// ---------------- WMMA tile loaders (per CDNA5 ISA layouts) ----------------
// A (16x32 f16, row-major src[ld]): lane<16 row m holds K {hi*16+0..15} (contiguous)
__device__ __forceinline__ v16h load_a_f16(const _Float16* src, int ld, int row0, int k0, int lane) {
    int m  = row0 + (lane & 15);
    int hi = (lane >> 4) * 16;
    const _Float16* p = src + (size_t)m * ld + k0;
    v16h a;
#pragma unroll
    for (int j = 0; j < 8; ++j) a[j] = p[hi + j];
#pragma unroll
    for (int j = 0; j < 8; ++j) a[8 + j] = p[hi + 8 + j];
    return a;
}

// A from an LDS-resident 16x32 f32 tile (row-major, stride 32)
__device__ __forceinline__ v16h load_a_lds(const float* p, int lane) {
    int m  = lane & 15;
    int hi = (lane >> 4) * 16;
    const float* row = p + m * 32;
    v16h a;
#pragma unroll
    for (int j = 0; j < 8; ++j) a[j] = (_Float16)row[hi + j];
#pragma unroll
    for (int j = 0; j < 8; ++j) a[8 + j] = (_Float16)row[hi + 8 + j];
    return a;
}

// B (32x16 f16) from K-major ("transposed") storage bt[n*ld + k]:
// lane n = column, halfs j -> K = (lane>=16 ? 16 : 0) + j  => contiguous per lane.
__device__ __forceinline__ v16h load_bt_f16(const _Float16* bt, int ld, int n0, int k0, int lane) {
    int n  = n0 + (lane & 15);
    int hi = (lane >> 4) * 16;
    const _Float16* p = bt + (size_t)n * ld + k0 + hi;
    v16h b;
#pragma unroll
    for (int j = 0; j < 16; ++j) b[j] = p[j];
    return b;
}

// B (32x16 f16) from row-major storage src[k*ld + n] (strided per lane).
__device__ __forceinline__ v16h load_b_rm(const _Float16* src, int ld, int k0, int n0, int lane) {
    int n = n0 + (lane & 15);
    int k = k0 + (lane >> 4) * 16;
    const _Float16* p = src + (size_t)k * ld + n;
    v16h b;
#pragma unroll
    for (int j = 0; j < 16; ++j) b[j] = p[(size_t)j * ld];
    return b;
}

// wave-half reductions (rows of a C tile live across 16 lanes of a half-wave)
__device__ __forceinline__ float hmax16(float v) {
#pragma unroll
    for (int off = 1; off < 16; off <<= 1) v = fmaxf(v, __shfl_xor(v, off, 32));
    return v;
}
__device__ __forceinline__ float hsum16(float v) {
#pragma unroll
    for (int off = 1; off < 16; off <<= 1) v += __shfl_xor(v, off, 32);
    return v;
}

// ---------------- elementwise / prep kernels ----------------
__global__ void k_f32_to_f16(const float* __restrict__ in, _Float16* __restrict__ out, int n) {
    int i = blockIdx.x * blockDim.x + threadIdx.x;
    if (i < n) out[i] = (_Float16)in[i];
}

// W[K x N] f32 -> WT[N x K] f16
__global__ void k_transpose_f16(const float* __restrict__ w, _Float16* __restrict__ wt, int K, int N) {
    int i = blockIdx.x * blockDim.x + threadIdx.x;
    if (i >= K * N) return;
    int k = i / N, n = i - k * N;
    wt[(size_t)n * K + k] = (_Float16)w[i];
}

// ---------------- WMMA GEMM: C[M,N] = A[M,K] * BT[N,K]^T + bias (+resid, relu) ----
// One wave owns a 16x64 supertile. N and K are compile-time so all tile bases
// fold into 24-bit instruction offsets (one A pointer + one BT pointer live).
// K-loop advances 64/iter with two ping-pong register sets (K % 64 == 0).
template <int N, int K>
__global__ __launch_bounds__(128, 2)
void k_gemm(const _Float16* __restrict__ A, const _Float16* __restrict__ BT,
            const float* __restrict__ bias, const float* __restrict__ resid,
            float* __restrict__ outF, _Float16* __restrict__ outH, int relu) {
    int lane = threadIdx.x & 31;
    int wv   = threadIdx.x >> 5;
    int tile = blockIdx.x * 4 + wv;
    constexpr int ntn = N >> 6;        // number of 64-wide supertiles
    int tm   = tile / ntn;
    int tn   = tile - tm * ntn;
    int row0 = tm << 4, col0 = tn << 6;

    // single live base pointers; all tile/slab deltas are constexpr immediates
    const _Float16* pa = A  + (size_t)(row0 + (lane & 15)) * K + (lane >> 4) * 16;
    const _Float16* pb = BT + (size_t)(col0 + (lane & 15)) * K + (lane >> 4) * 16;

    v8f cc[4] = {{}, {}, {}, {}};

    auto lda = [&](int k0) {
        v16h a;
#pragma unroll
        for (int j = 0; j < 8; ++j) a[j] = pa[k0 + j];
#pragma unroll
        for (int j = 0; j < 8; ++j) a[8 + j] = pa[k0 + 8 + j];
        return a;
    };
    auto ldb = [&](int t, int k0) {
        v16h b;
#pragma unroll
        for (int j = 0; j < 16; ++j) b[j] = pb[(size_t)t * 16 * K + k0 + j];
        return b;
    };

    // even-slab register set, preloaded at k=0
    v16h ea  = lda(0);
    v16h eb0 = ldb(0, 0), eb1 = ldb(1, 0), eb2 = ldb(2, 0), eb3 = ldb(3, 0);

    for (int k0 = 0; k0 < K; k0 += 64) {
        int km = k0 + 32;              // odd slab (always < K since K % 64 == 0)
        v16h oa  = lda(km);
        v16h ob0 = ldb(0, km), ob1 = ldb(1, km), ob2 = ldb(2, km), ob3 = ldb(3, km);
        __builtin_prefetch((const void*)(pa + km + 32), 0, 1);

        cc[0] = WMMA16(ea, eb0, cc[0]);
        cc[1] = WMMA16(ea, eb1, cc[1]);
        cc[2] = WMMA16(ea, eb2, cc[2]);
        cc[3] = WMMA16(ea, eb3, cc[3]);

        int ke = k0 + 64;
        if (ke < K) {
            ea  = lda(ke);
            eb0 = ldb(0, ke); eb1 = ldb(1, ke); eb2 = ldb(2, ke); eb3 = ldb(3, ke);
        }
        cc[0] = WMMA16(oa, ob0, cc[0]);
        cc[1] = WMMA16(oa, ob1, cc[1]);
        cc[2] = WMMA16(oa, ob2, cc[2]);
        cc[3] = WMMA16(oa, ob3, cc[3]);
    }

    int nn = lane & 15;
    int mb = row0 + ((lane >> 4) << 3);
#pragma unroll
    for (int t = 0; t < 4; ++t) {
        int n = col0 + t * 16 + nn;
        float bv = bias ? bias[n] : 0.0f;
#pragma unroll
        for (int r = 0; r < 8; ++r) {
            size_t idx = (size_t)(mb + r) * N + n;
            float y = cc[t][r] + bv;
            if (resid) y += resid[idx];
            if (relu)  y = fmaxf(y, 0.0f);
            if (outF) outF[idx] = y;
            if (outH) outH[idx] = (_Float16)y;
        }
    }
}

// ---------------- fused dual-softmax flash attention ----------------
// one wave = one (b, h, 16-query tile); dual online softmax (global + local window |i-j|<=4)
__global__ __launch_bounds__(128, 1)
void k_attn(const _Float16* __restrict__ Qh, const _Float16* __restrict__ Kh,
            const _Float16* __restrict__ Vh,
            const float* __restrict__ adj_w, const float* __restrict__ adj_b,
            float* __restrict__ glb, float* __restrict__ lcl) {
    __shared__ float lds[4][16 * 32];
    int lane = threadIdx.x & 31;
    int wv   = threadIdx.x >> 5;
    int wid  = blockIdx.x * 4 + wv;
    int qt   = wid & (Ss / 16 - 1);
    int bh   = wid >> 7;
    int h    = bh & (Hh - 1);
    int b    = bh >> 2;

    const _Float16* Qbase = Qh + (size_t)b * Ss * HIDc + h * DHh;
    const _Float16* Kbase = Kh + (size_t)b * Ss * HIDc + h * DHh;
    const _Float16* Vbase = Vh + (size_t)b * Ss * HIDc + h * DHh;

    int q0 = qt << 4;
    const float scale = 0.0625f;            // 1/sqrt(HID)
    const float aw = adj_w[0], ab = adj_b[0];
    const float NEG = -1e9f;

    // Q as two A tiles (d = 0..31, 32..63); Qbase rows are sequence positions
    v16h qa0 = load_a_f16(Qbase, HIDc, q0, 0,  lane);
    v16h qa1 = load_a_f16(Qbase, HIDc, q0, 32, lane);

    v8f accG0 = {}, accG1 = {}, accG2 = {}, accG3 = {};
    v8f accL0 = {}, accL1 = {}, accL2 = {}, accL3 = {};
    float mG[8], lG[8], mL[8], lL[8];
#pragma unroll
    for (int r = 0; r < 8; ++r) { mG[r] = -1e30f; lG[r] = 0.f; mL[r] = -1e30f; lL[r] = 0.f; }

    int ncol = lane & 15;
    int mb   = (lane >> 4) << 3;

    for (int kb = 0; kb < Ss / 32; ++kb) {
        int key0 = kb << 5;

        // scores: Q(16x64) x K^T(64x32) -> two 16x16 tiles
        v8f s0 = {}, s1 = {};
        {
            v16h bk = load_bt_f16(Kbase, HIDc, key0, 0, lane);
            s0 = WMMA16(qa0, bk, s0);
            bk = load_bt_f16(Kbase, HIDc, key0, 32, lane);
            s0 = WMMA16(qa1, bk, s0);
            bk = load_bt_f16(Kbase, HIDc, key0 + 16, 0, lane);
            s1 = WMMA16(qa0, bk, s1);
            bk = load_bt_f16(Kbase, HIDc, key0 + 16, 32, lane);
            s1 = WMMA16(qa1, bk, s1);
        }

        float j0 = (float)(key0 + ncol);
        float j1 = j0 + 16.0f;
        float pl0[8], pl1[8];
#pragma unroll
        for (int r = 0; r < 8; ++r) {
            float iq = (float)(q0 + mb + r);
            float d0 = fabsf(iq - j0), d1 = fabsf(iq - j1);
            float sg0 = s0[r] * scale + __expf(-fabsf(aw * d0 * d0 - ab));
            float sg1 = s1[r] * scale + __expf(-fabsf(aw * d1 * d1 - ab));
            float sl0 = (d0 <= 4.0f) ? sg0 : NEG;
            float sl1 = (d1 <= 4.0f) ? sg1 : NEG;

            // global online softmax
            float bm  = hmax16(fmaxf(sg0, sg1));
            float mn  = fmaxf(mG[r], bm);
            float scl = __expf(mG[r] - mn);
            float p0  = __expf(sg0 - mn), p1 = __expf(sg1 - mn);
            lG[r] = lG[r] * scl + hsum16(p0 + p1);
            mG[r] = mn;
            accG0[r] *= scl; accG1[r] *= scl; accG2[r] *= scl; accG3[r] *= scl;
            lds[wv][(mb + r) * 32 + ncol]      = p0;
            lds[wv][(mb + r) * 32 + 16 + ncol] = p1;

            // local online softmax
            bm  = hmax16(fmaxf(sl0, sl1));
            mn  = fmaxf(mL[r], bm);
            scl = __expf(mL[r] - mn);
            p0  = __expf(sl0 - mn); p1 = __expf(sl1 - mn);
            lL[r] = lL[r] * scl + hsum16(p0 + p1);
            mL[r] = mn;
            accL0[r] *= scl; accL1[r] *= scl; accL2[r] *= scl; accL3[r] *= scl;
            pl0[r] = p0; pl1[r] = p1;
        }

        // V as four B tiles over d-groups (K dim = 32 keys); loaded late to
        // keep them out of the softmax section's live range.
        v16h vb0 = load_b_rm(Vbase, HIDc, key0, 0,  lane);
        v16h vb1 = load_b_rm(Vbase, HIDc, key0, 16, lane);
        v16h vb2 = load_b_rm(Vbase, HIDc, key0, 32, lane);
        v16h vb3 = load_b_rm(Vbase, HIDc, key0, 48, lane);

        // global P (C-layout in LDS) -> A-layout, P(16x32) x V(32x64)
        v16h pa = load_a_lds(&lds[wv][0], lane);
        accG0 = WMMA16(pa, vb0, accG0);
        accG1 = WMMA16(pa, vb1, accG1);
        accG2 = WMMA16(pa, vb2, accG2);
        accG3 = WMMA16(pa, vb3, accG3);

#pragma unroll
        for (int r = 0; r < 8; ++r) {
            lds[wv][(mb + r) * 32 + ncol]      = pl0[r];
            lds[wv][(mb + r) * 32 + 16 + ncol] = pl1[r];
        }
        pa = load_a_lds(&lds[wv][0], lane);
        accL0 = WMMA16(pa, vb0, accL0);
        accL1 = WMMA16(pa, vb1, accL1);
        accL2 = WMMA16(pa, vb2, accL2);
        accL3 = WMMA16(pa, vb3, accL3);
    }

    // epilogue: divide by softmax sums, scatter to [b*S+s][h*64+d]
#pragma unroll
    for (int r = 0; r < 8; ++r) {
        size_t row = (size_t)b * Ss + q0 + mb + r;
        size_t base = row * HIDc + h * DHh + ncol;
        float ig = 1.0f / lG[r], il = 1.0f / lL[r];
        glb[base +  0] = accG0[r] * ig;
        glb[base + 16] = accG1[r] * ig;
        glb[base + 32] = accG2[r] * ig;
        glb[base + 48] = accG3[r] * ig;
        lcl[base +  0] = accL0[r] * il;
        lcl[base + 16] = accL1[r] * il;
        lcl[base + 32] = accL2[r] * il;
        lcl[base + 48] = accL3[r] * il;
    }
}

// tmp = sigmoid(alpha)*glb + (1-a)*lcl, signed-sqrt; written in place over glb
__global__ void k_combine(const float* __restrict__ lclp, float* __restrict__ glb_tmp,
                          const float* __restrict__ alpha, int n) {
    int i = blockIdx.x * blockDim.x + threadIdx.x;
    if (i >= n) return;
    float a = 1.0f / (1.0f + __expf(-alpha[0]));
    float t = a * glb_tmp[i] + (1.0f - a) * lclp[i];
    float r = (t > 0.f) ? sqrtf(t) : ((t < 0.f) ? -sqrtf(-t) : 0.f);
    glb_tmp[i] = r;
}

// L2 norm along the sequence axis: one block per (b, channel)
__global__ void k_colnorm(const float* __restrict__ tmp, float* __restrict__ nrm) {
    __shared__ float red[256];
    int c = blockIdx.x & (HIDc - 1);
    int b = blockIdx.x >> 8;
    int tid = threadIdx.x;
    float s = 0.f;
    for (int sidx = tid; sidx < Ss; sidx += 256) {
        float v = tmp[((size_t)b * Ss + sidx) * HIDc + c];
        s += v * v;
    }
    red[tid] = s; __syncthreads();
    for (int st = 128; st > 0; st >>= 1) {
        if (tid < st) red[tid] += red[tid + st];
        __syncthreads();
    }
    if (tid == 0) nrm[b * HIDc + c] = fmaxf(sqrtf(red[0]), 1e-12f);
}

__global__ void k_normalize(const float* __restrict__ tmp, const float* __restrict__ nrm,
                            _Float16* __restrict__ outh, int n) {
    int i = blockIdx.x * blockDim.x + threadIdx.x;
    if (i >= n) return;
    int c = i & (HIDc - 1);
    int b = i / (Ss * HIDc);
    outh[i] = (_Float16)(tmp[i] / nrm[b * HIDc + c]);
}

// LayerNorm over DM, one block (256 threads) per row
__global__ void k_layernorm(const float* __restrict__ x, const float* __restrict__ g,
                            const float* __restrict__ be,
                            float* __restrict__ outF, _Float16* __restrict__ outH) {
    __shared__ float red[256];
    int row = blockIdx.x, tid = threadIdx.x;
    const float* p = x + (size_t)row * DM;
    float s = 0.f;
    for (int i = tid; i < DM; i += 256) s += p[i];
    red[tid] = s; __syncthreads();
    for (int st = 128; st > 0; st >>= 1) { if (tid < st) red[tid] += red[tid + st]; __syncthreads(); }
    float mu = red[0] / DM;
    __syncthreads();
    float v = 0.f;
    for (int i = tid; i < DM; i += 256) { float d = p[i] - mu; v += d * d; }
    red[tid] = v; __syncthreads();
    for (int st = 128; st > 0; st >>= 1) { if (tid < st) red[tid] += red[tid + st]; __syncthreads(); }
    float inv = rsqrtf(red[0] / DM + 1e-5f);
    for (int i = tid; i < DM; i += 256) {
        float y = (p[i] - mu) * inv * g[i] + be[i];
        size_t idx = (size_t)row * DM + i;
        if (outF) outF[idx] = y;
        if (outH) outH[idx] = (_Float16)y;
    }
}

// ---------------- workspace layout (bytes, phase-aliased) ----------------
constexpr size_t O_QH16 = 0;                          //  8 MB  q f16            (later: xa / x2 low half)
constexpr size_t O_KH16 = O_QH16 + (size_t)Mrows * DM * 2;   // 8 MB  (xa/x2 hi)
constexpr size_t O_VH16 = O_KH16 + (size_t)Mrows * DM * 2;   // 8 MB  (later: x1h)
constexpr size_t O_WQT  = O_VH16 + (size_t)Mrows * DM * 2;
constexpr size_t O_WKT  = O_WQT + (size_t)HIDc * DM * 2;
constexpr size_t O_WVT  = O_WKT + (size_t)HIDc * DM * 2;
constexpr size_t O_WOT  = O_WVT + (size_t)HIDc * DM * 2;
constexpr size_t O_W1T  = O_WOT + (size_t)DM * HIDc * 2;
constexpr size_t O_W2T  = O_W1T + (size_t)DFF * DM * 2;
constexpr size_t O_QHP  = O_W2T + (size_t)DM * DFF * 2;      // 2 MB  (later: tmpn f16)
constexpr size_t O_KHP  = O_QHP + (size_t)Mrows * HIDc * 2;
constexpr size_t O_VHP  = O_KHP + (size_t)Mrows * HIDc * 2;
constexpr size_t O_GLB  = O_VHP + (size_t)Mrows * HIDc * 2;  // 4 MB  (tmp in place)
constexpr size_t O_LCL  = O_GLB + (size_t)Mrows * HIDc * 4;  // 4 MB  (later: ffn hidden f16)
constexpr size_t O_X1   = O_LCL + (size_t)Mrows * HIDc * 4;  // 16 MB
constexpr size_t O_NRM  = O_X1 + (size_t)Mrows * DM * 4;

extern "C" void kernel_launch(void* const* d_in, const int* in_sizes, int n_in,
                              void* d_out, int out_size, void* d_ws, size_t ws_size,
                              hipStream_t stream) {
    const float* q    = (const float*)d_in[0];
    const float* k    = (const float*)d_in[1];
    const float* v    = (const float*)d_in[2];
    const float* Wq   = (const float*)d_in[3];
    const float* bq   = (const float*)d_in[4];
    const float* Wk   = (const float*)d_in[5];
    const float* bk   = (const float*)d_in[6];
    const float* Wv   = (const float*)d_in[7];
    const float* bv   = (const float*)d_in[8];
    const float* Wo   = (const float*)d_in[9];
    const float* bo   = (const float*)d_in[10];
    const float* alpha= (const float*)d_in[11];
    const float* adjw = (const float*)d_in[12];
    const float* adjb = (const float*)d_in[13];
    const float* W1   = (const float*)d_in[14];
    const float* b1   = (const float*)d_in[15];
    const float* W2   = (const float*)d_in[16];
    const float* b2   = (const float*)d_in[17];
    const float* g1   = (const float*)d_in[18];
    const float* be1  = (const float*)d_in[19];
    const float* g2   = (const float*)d_in[20];
    const float* be2  = (const float*)d_in[21];

    char* ws = (char*)d_ws;
    _Float16* qh16 = (_Float16*)(ws + O_QH16);
    _Float16* kh16 = (_Float16*)(ws + O_KH16);
    _Float16* vh16 = (_Float16*)(ws + O_VH16);
    _Float16* WqT  = (_Float16*)(ws + O_WQT);
    _Float16* WkT  = (_Float16*)(ws + O_WKT);
    _Float16* WvT  = (_Float16*)(ws + O_WVT);
    _Float16* WoT  = (_Float16*)(ws + O_WOT);
    _Float16* W1T  = (_Float16*)(ws + O_W1T);
    _Float16* W2T  = (_Float16*)(ws + O_W2T);
    _Float16* Qhp  = (_Float16*)(ws + O_QHP);
    _Float16* Khp  = (_Float16*)(ws + O_KHP);
    _Float16* Vhp  = (_Float16*)(ws + O_VHP);
    float*    glb  = (float*)(ws + O_GLB);
    float*    lcl  = (float*)(ws + O_LCL);
    float*    x1   = (float*)(ws + O_X1);
    float*    nrm  = (float*)(ws + O_NRM);
    // phase aliases
    float*    xa    = (float*)(ws + O_QH16);   // 16 MB over qh16+kh16
    float*    x2    = (float*)(ws + O_QH16);   // reused after LN1 consumes xa
    _Float16* x1h   = (_Float16*)(ws + O_VH16);
    _Float16* hh    = (_Float16*)(ws + O_LCL);
    _Float16* tmpnh = (_Float16*)(ws + O_QHP);

    dim3 blk(128);
    int nQKV = Mrows * DM;

    // 1) q/k/v -> f16
    k_f32_to_f16<<<nQKV / 256, 256, 0, stream>>>(q, qh16, nQKV);
    k_f32_to_f16<<<nQKV / 256, 256, 0, stream>>>(k, kh16, nQKV);
    k_f32_to_f16<<<nQKV / 256, 256, 0, stream>>>(v, vh16, nQKV);

    // 2) weights -> transposed f16
    k_transpose_f16<<<(DM * HIDc) / 256, 256, 0, stream>>>(Wq, WqT, DM, HIDc);
    k_transpose_f16<<<(DM * HIDc) / 256, 256, 0, stream>>>(Wk, WkT, DM, HIDc);
    k_transpose_f16<<<(DM * HIDc) / 256, 256, 0, stream>>>(Wv, WvT, DM, HIDc);
    k_transpose_f16<<<(HIDc * DM) / 256, 256, 0, stream>>>(Wo, WoT, HIDc, DM);
    k_transpose_f16<<<(DM * DFF) / 256, 256, 0, stream>>>(W1, W1T, DM, DFF);
    k_transpose_f16<<<(DFF * DM) / 256, 256, 0, stream>>>(W2, W2T, DFF, DM);

    // 3) QKV projections (f16 out); supertiles = (M/16)*(N/64)
    int st_qkv = (Mrows / 16) * (HIDc / 64);
    k_gemm<HIDc, DM><<<st_qkv / 4, blk, 0, stream>>>(qh16, WqT, bq, nullptr, nullptr, Qhp, 0);
    k_gemm<HIDc, DM><<<st_qkv / 4, blk, 0, stream>>>(kh16, WkT, bk, nullptr, nullptr, Khp, 0);
    k_gemm<HIDc, DM><<<st_qkv / 4, blk, 0, stream>>>(vh16, WvT, bv, nullptr, nullptr, Vhp, 0);

    // 4) fused dual-softmax flash attention
    k_attn<<<(Bb * Hh * (Ss / 16)) / 4, blk, 0, stream>>>(Qhp, Khp, Vhp, adjw, adjb, glb, lcl);

    // 5) gated combine + signed sqrt (in place over glb)
    int nTmp = Mrows * HIDc;
    k_combine<<<nTmp / 256, 256, 0, stream>>>(lcl, glb, alpha, nTmp);

    // 6) L2 norm over sequence axis
    k_colnorm<<<Bb * HIDc, 256, 0, stream>>>(glb, nrm);

    // 7) normalize -> f16
    k_normalize<<<nTmp / 256, 256, 0, stream>>>(glb, nrm, tmpnh, nTmp);

    // 8) output projection + residual(q)
    int st_dm = (Mrows / 16) * (DM / 64);
    k_gemm<DM, HIDc><<<st_dm / 4, blk, 0, stream>>>(tmpnh, WoT, bo, q, xa, nullptr, 0);

    // 9) LayerNorm 1 -> x1 (f32 + f16)
    k_layernorm<<<Mrows, 256, 0, stream>>>(xa, g1, be1, x1, x1h);

    // 10) FFN up + ReLU (f16 out)
    int st_ff = (Mrows / 16) * (DFF / 64);
    k_gemm<DFF, DM><<<st_ff / 4, blk, 0, stream>>>(x1h, W1T, b1, nullptr, nullptr, hh, 1);

    // 11) FFN down + residual(x1)
    k_gemm<DM, DFF><<<st_dm / 4, blk, 0, stream>>>(hh, W2T, b2, x1, x2, nullptr, 0);

    // 12) LayerNorm 2 -> final output
    k_layernorm<<<Mrows, 256, 0, stream>>>(x2, g2, be2, (float*)d_out, nullptr);
}
